// GeneralizedHamiltonianDynamics_89464168775889
// MI455X (gfx1250) — matrix operand order repaired and checked by reading, not compiled, not run on hardware
//
#include <hip/hip_runtime.h>

#define BATCH 32768
#define DIN 64
#define HID 1024

typedef __attribute__((ext_vector_type(16))) __bf16 v16bf;
typedef __attribute__((ext_vector_type(8)))  __bf16 v8bf;
typedef __attribute__((ext_vector_type(8)))  float  v8f;

// ---------------------------------------------------------------------------
// Fast tanh: hardware v_tanh_f32 on gfx1250 if the builtin exists, otherwise
// a branch-free v_exp_f32 + v_rcp_f32 sequence (no exec-mask churn).
// ---------------------------------------------------------------------------
__device__ __forceinline__ float fast_tanh(float x) {
#if __has_builtin(__builtin_amdgcn_tanhf)
    return __builtin_amdgcn_tanhf(x);
#else
    float xc = fminf(fmaxf(x, -15.0f), 15.0f);
    float e  = __builtin_amdgcn_exp2f(xc * 2.8853900817779268f); // 2*log2(e)
    return 1.0f - 2.0f * __builtin_amdgcn_rcpf(e + 1.0f);
#endif
}

// ---------------------------------------------------------------------------
// WMMA wrapper: D = A(16x32 bf16) x B(32x16 bf16) + C(16x16 f32)
// ---------------------------------------------------------------------------
__device__ __forceinline__ v8f wmma_bf16(v16bf a, v16bf b, v8f c) {
    return __builtin_amdgcn_wmma_f32_16x16x32_bf16(
        false, a, false, b, (short)0, c, false, false);
}

// A-fragment (16x32 bf16) from LDS row-major [16][stride]:
// lane m = lane&15, hi = lane>>4; e<8 -> K=k0+hi*8+e, e>=8 -> K=k0+16+hi*8+(e-8)
__device__ __forceinline__ v16bf a_frag_lds(const __bf16* buf, int stride,
                                            int m, int k0, int hi) {
    union { v16bf v; v8bf h[2]; } u;
    const __bf16* p = buf + m * stride + k0 + hi * 8;
    u.h[0] = *reinterpret_cast<const v8bf*>(p);
    u.h[1] = *reinterpret_cast<const v8bf*>(p + 16);
    return u.v;
}

// B-fragment (32x16 bf16) from column-major weights w[n*Ktot + k]:
// e<8 -> K=k0+hi*8+e, e>=8 -> K=k0+16+hi*8+(e-8) (two aligned 16B loads)
__device__ __forceinline__ v16bf b_frag(const __bf16* w, int Ktot,
                                        int n, int k0, int hi) {
    union { v16bf v; v8bf h[2]; } u;
    const __bf16* p = w + (size_t)n * Ktot + k0 + hi * 8;
    u.h[0] = *reinterpret_cast<const v8bf*>(p);
    u.h[1] = *reinterpret_cast<const v8bf*>(p + 16);
    return u.v;
}

// ---------------------------------------------------------------------------
// Prep: f32 -> bf16 weight convert, optional transpose
// ---------------------------------------------------------------------------
__global__ void convert_tr_kernel(const float* __restrict__ src,
                                  __bf16* __restrict__ dst,
                                  int R, int C, int transpose) {
    int idx = blockIdx.x * blockDim.x + threadIdx.x;
    if (idx >= R * C) return;
    int r = idx / C, c = idx % C;
    __bf16 v = (__bf16)src[idx];
    if (transpose) dst[c * R + r] = v;
    else           dst[idx] = v;
}

// ---------------------------------------------------------------------------
// Fused HNN kernel: 4 waves (128 threads) cooperate on one 16-row batch tile.
// Wave w owns column tiles [w*16, w*16+16) of the 1024-wide phases,
// processed two at a time (two independent WMMA accumulator chains).
// ---------------------------------------------------------------------------
__global__ __launch_bounds__(128)
void hnn_fused_kernel(const float* __restrict__ z,
                      const float* __restrict__ b1,
                      const float* __restrict__ b2,
                      const float* __restrict__ W3,
                      const float* __restrict__ bf1,
                      const float* __restrict__ bf2,
                      const __bf16* __restrict__ W1cm,   // [HID][DIN]  (= W1^T)
                      const __bf16* __restrict__ W1rm,   // [DIN][HID]  (= W1)
                      const __bf16* __restrict__ W2cm,   // [HID][HID]  (= W2^T)
                      const __bf16* __restrict__ W2rm,   // [HID][HID]  (= W2)
                      const __bf16* __restrict__ Wf1cm,  // [HID][DIN]  (= Wf1^T)
                      const __bf16* __restrict__ Wf2cm,  // [DIN][HID]  (= Wf2^T)
                      float* __restrict__ out) {
    __shared__ __bf16 h1buf[16 * HID];   // h1, later overwritten in place by g1
    __shared__ __bf16 g2buf[16 * HID];   // g2, later reused for f1

    const int lane = threadIdx.x & 31;
    const int wid  = threadIdx.x >> 5;   // 0..3
    const int nL   = lane & 15;
    const int hi   = lane >> 4;
    const int m    = nL;                 // A row within tile
    const int m0   = blockIdx.x * 16;
    const int ntLo = wid * 16;           // this wave's column-tile range
    const int ntHi = ntLo + 16;

    // ---- z A-fragments (K = 64 -> two k-blocks), registers ----------------
    v16bf za[2];
    {
        const float* zrow = z + (size_t)(m0 + m) * DIN;
        #pragma unroll
        for (int kb = 0; kb < 2; ++kb) {
            union { v16bf v; __bf16 e[16]; } u;
            const int k0 = kb * 32;
            #pragma unroll
            for (int e = 0; e < 8; ++e)
                u.e[e] = (__bf16)zrow[k0 + hi * 8 + e];
            #pragma unroll
            for (int e = 0; e < 8; ++e)
                u.e[8 + e] = (__bf16)zrow[k0 + 16 + hi * 8 + e];
            za[kb] = u.v;
        }
    }

    // ---- layer 1: h1 = tanh(z @ W1 + b1) ----------------------------------
    for (int nt = ntLo; nt < ntHi; nt += 2) {
        const int n0 = nt * 16 + nL, n1 = n0 + 16;
        v8f acc0, acc1;
        {
            const float bi0 = b1[n0], bi1 = b1[n1];
            #pragma unroll
            for (int r = 0; r < 8; ++r) { acc0[r] = bi0; acc1[r] = bi1; }
        }
        #pragma unroll
        for (int kb = 0; kb < 2; ++kb) {
            acc0 = wmma_bf16(za[kb], b_frag(W1cm, DIN, n0, kb * 32, hi), acc0);
            acc1 = wmma_bf16(za[kb], b_frag(W1cm, DIN, n1, kb * 32, hi), acc1);
        }
        #pragma unroll
        for (int r = 0; r < 8; ++r) {
            h1buf[(r + 8 * hi) * HID + n0] = (__bf16)fast_tanh(acc0[r]);
            h1buf[(r + 8 * hi) * HID + n1] = (__bf16)fast_tanh(acc1[r]);
        }
    }
    __syncthreads();

    // ---- layer 2: g2 = W3 * (1 - tanh(h1@W2 + b2)^2) ----------------------
    for (int nt = ntLo; nt < ntHi; nt += 2) {
        const int n0 = nt * 16 + nL, n1 = n0 + 16;
        v8f acc0, acc1;
        {
            const float bi0 = b2[n0], bi1 = b2[n1];
            #pragma unroll
            for (int r = 0; r < 8; ++r) { acc0[r] = bi0; acc1[r] = bi1; }
        }
        #pragma unroll 4
        for (int k0 = 0; k0 < HID; k0 += 32) {
            v16bf a = a_frag_lds(h1buf, HID, m, k0, hi);
            acc0 = wmma_bf16(a, b_frag(W2cm, HID, n0, k0, hi), acc0);
            acc1 = wmma_bf16(a, b_frag(W2cm, HID, n1, k0, hi), acc1);
        }
        const float w30 = W3[n0], w31 = W3[n1];
        #pragma unroll
        for (int r = 0; r < 8; ++r) {
            float h20 = fast_tanh(acc0[r]);
            float h21 = fast_tanh(acc1[r]);
            g2buf[(r + 8 * hi) * HID + n0] = (__bf16)(w30 * (1.0f - h20 * h20));
            g2buf[(r + 8 * hi) * HID + n1] = (__bf16)(w31 * (1.0f - h21 * h21));
        }
    }
    __syncthreads();

    // ---- backward: g1 = (g2 @ W2^T) * (1 - h1^2), overwrite h1buf ---------
    for (int nt = ntLo; nt < ntHi; nt += 2) {
        const int n0 = nt * 16 + nL, n1 = n0 + 16;
        v8f acc0, acc1;
        #pragma unroll
        for (int r = 0; r < 8; ++r) { acc0[r] = 0.0f; acc1[r] = 0.0f; }
        #pragma unroll 4
        for (int k0 = 0; k0 < HID; k0 += 32) {
            v16bf a = a_frag_lds(g2buf, HID, m, k0, hi);
            acc0 = wmma_bf16(a, b_frag(W2rm, HID, n0, k0, hi), acc0);
            acc1 = wmma_bf16(a, b_frag(W2rm, HID, n1, k0, hi), acc1);
        }
        #pragma unroll
        for (int r = 0; r < 8; ++r) {
            const int i0 = (r + 8 * hi) * HID + n0;
            const int i1 = (r + 8 * hi) * HID + n1;
            float h10 = (float)h1buf[i0], h11 = (float)h1buf[i1];
            h1buf[i0] = (__bf16)(acc0[r] * (1.0f - h10 * h10));
            h1buf[i1] = (__bf16)(acc1[r] * (1.0f - h11 * h11));
        }
    }
    __syncthreads();

    // ---- gradH tile needed by this wave's output tile ---------------------
    // out tile `wid` (cols 16*wid..) needs gradH tile src = wid<2 ? wid+2 : wid-2
    const int srcT = (wid < 2) ? wid + 2 : wid - 2;
    float gH[8];
    {
        const int n = srcT * 16 + nL;
        v8f acc;
        #pragma unroll
        for (int r = 0; r < 8; ++r) acc[r] = 0.0f;
        #pragma unroll 4
        for (int k0 = 0; k0 < HID; k0 += 32)
            acc = wmma_bf16(a_frag_lds(h1buf, HID, m, k0, hi),
                            b_frag(W1rm, HID, n, k0, hi), acc);
        #pragma unroll
        for (int r = 0; r < 8; ++r) gH[r] = acc[r];
    }

    // ---- forcing hidden: f1 = tanh(z @ Wf1 + bf1), reuse g2buf ------------
    for (int nt = ntLo; nt < ntHi; nt += 2) {
        const int n0 = nt * 16 + nL, n1 = n0 + 16;
        v8f acc0, acc1;
        {
            const float bi0 = bf1[n0], bi1 = bf1[n1];
            #pragma unroll
            for (int r = 0; r < 8; ++r) { acc0[r] = bi0; acc1[r] = bi1; }
        }
        #pragma unroll
        for (int kb = 0; kb < 2; ++kb) {
            acc0 = wmma_bf16(za[kb], b_frag(Wf1cm, DIN, n0, kb * 32, hi), acc0);
            acc1 = wmma_bf16(za[kb], b_frag(Wf1cm, DIN, n1, kb * 32, hi), acc1);
        }
        #pragma unroll
        for (int r = 0; r < 8; ++r) {
            g2buf[(r + 8 * hi) * HID + n0] = (__bf16)fast_tanh(acc0[r]);
            g2buf[(r + 8 * hi) * HID + n1] = (__bf16)fast_tanh(acc1[r]);
        }
    }
    __syncthreads();

    // ---- forcing out tile `wid`: fo = f1 @ Wf2 + bf2 ----------------------
    float fo[8];
    {
        const int n = wid * 16 + nL;
        const float bias = bf2[n];
        v8f acc;
        #pragma unroll
        for (int r = 0; r < 8; ++r) acc[r] = bias;
        #pragma unroll 4
        for (int k0 = 0; k0 < HID; k0 += 32)
            acc = wmma_bf16(a_frag_lds(g2buf, HID, m, k0, hi),
                            b_frag(Wf2cm, HID, n, k0, hi), acc);
        #pragma unroll
        for (int r = 0; r < 8; ++r) fo[r] = acc[r];
    }

    // ---- symplectic combine + store ---------------------------------------
    {
        const float sgn = (wid < 2) ? 1.0f : -1.0f;
        const int   n   = wid * 16 + nL;
        #pragma unroll
        for (int r = 0; r < 8; ++r)
            out[(size_t)(m0 + r + 8 * hi) * DIN + n] = sgn * gH[r] + fo[r];
    }
}

// ---------------------------------------------------------------------------
extern "C" void kernel_launch(void* const* d_in, const int* in_sizes, int n_in,
                              void* d_out, int out_size, void* d_ws, size_t ws_size,
                              hipStream_t stream) {
    (void)in_sizes; (void)n_in; (void)out_size; (void)ws_size;
    const float* z   = (const float*)d_in[0];
    const float* W1  = (const float*)d_in[1];
    const float* b1  = (const float*)d_in[2];
    const float* W2  = (const float*)d_in[3];
    const float* b2  = (const float*)d_in[4];
    const float* W3  = (const float*)d_in[5];
    /* b3 = d_in[6]: no effect on gradH, unused */
    const float* Wf1 = (const float*)d_in[7];
    const float* bf1 = (const float*)d_in[8];
    const float* Wf2 = (const float*)d_in[9];
    const float* bf2 = (const float*)d_in[10];
    float* out = (float*)d_out;

    __bf16* ws    = (__bf16*)d_ws;
    __bf16* W1cm  = ws;                        // 1024*64
    __bf16* W1rm  = ws + 65536;                // 64*1024
    __bf16* W2cm  = ws + 131072;               // 1024*1024
    __bf16* W2rm  = ws + 131072 + 1048576;     // 1024*1024
    __bf16* Wf1cm = ws + 131072 + 2097152;     // 1024*64
    __bf16* Wf2cm = Wf1cm + 65536;             // 64*1024

    auto conv = [&](const float* s, __bf16* d, int R, int C, int tr) {
        int n = R * C;
        convert_tr_kernel<<<(n + 255) / 256, 256, 0, stream>>>(s, d, R, C, tr);
    };
    conv(W1,  W1cm,  DIN, HID, 1);
    conv(W1,  W1rm,  DIN, HID, 0);
    conv(W2,  W2cm,  HID, HID, 1);
    conv(W2,  W2rm,  HID, HID, 0);
    conv(Wf1, Wf1cm, DIN, HID, 1);
    conv(Wf2, Wf2cm, HID, DIN, 1);

    hnn_fused_kernel<<<BATCH / 16, 128, 0, stream>>>(
        z, b1, b2, W3, bf1, bf2,
        W1cm, W1rm, W2cm, W2rm, Wf1cm, Wf2cm, out);
}
